// ConvS2SAny2Many_47029891891741
// MI455X (gfx1250) — compile-verified, matrix operand order checked by reading, not run on hardware
//
#include <hip/hip_runtime.h>
#include <hip/hip_bf16.h>
#include <math.h>

// ---------- CDNA5 WMMA types ----------
typedef __attribute__((ext_vector_type(16))) __bf16    v16bf;
typedef __attribute__((ext_vector_type(8)))  float     v8f;
typedef __attribute__((ext_vector_type(8)))  unsigned  v8u;
typedef __attribute__((ext_vector_type(4)))  int       v4i;

typedef __attribute__((address_space(1))) v4i as1_v4i;
typedef __attribute__((address_space(3))) v4i as3_v4i;

#if __has_builtin(__builtin_amdgcn_global_load_async_to_lds_b128) && \
    __has_builtin(__builtin_amdgcn_s_wait_asynccnt)
#define HAVE_ASYNC_LDS 1
#else
#define HAVE_ASYNC_LDS 0
#endif

#define SQRT_HALF_F 0.70710678118654752f

static __device__ __forceinline__ unsigned short f2bf(float x) {
  unsigned u = __builtin_bit_cast(unsigned, x);
  u += 0x7FFFu + ((u >> 16) & 1u);           // round-to-nearest-even
  return (unsigned short)(u >> 16);
}

// 16B global->LDS copy: async (ASYNCcnt) when available, vector copy otherwise
static __device__ __forceinline__ void copy_b128_to_lds(const void* g, void* lds) {
#if HAVE_ASYNC_LDS
  __builtin_amdgcn_global_load_async_to_lds_b128(
      (as1_v4i*)(unsigned long long)g,
      (as3_v4i*)(unsigned)(unsigned long long)lds, 0, 0);
#else
  *(uint4*)lds = *(const uint4*)g;
#endif
}

static __device__ __forceinline__ void staging_wait() {
#if HAVE_ASYNC_LDS
  __builtin_amdgcn_s_wait_asynccnt(0);
#endif
}

// =====================================================================
// Generic batched GEMM:  C[M,N] = alpha * op(A)[M,K] * B[K,N] + bias[M]
//   A: bf16, row-major [M,lda] or col-major (elem (m,k) = A[k*lda+m])
//   B: bf16 [K,ldb] or f32 [K,ldb] (converted during LDS staging)
//   C: optional f32 out; Cbf: optional bf16 out
// Tile: 128x128 per 256-thread block (8 waves, each 32x64 -> 2x4 WMMA accums)
// =====================================================================
__global__ __launch_bounds__(256) void gemm_wmma(
    const unsigned short* __restrict__ A,
    const void* __restrict__ Bsrc,
    const float* __restrict__ bias,
    float* __restrict__ C,
    unsigned short* __restrict__ Cbf,
    int M, int N, int K,
    int lda, int ldb, int ldc,
    long long strideA, long long strideB, long long strideC,
    float alpha, int a_colmajor, int b_is_f32)
{
  __shared__ alignas(16) unsigned short As[128 * 32];   // [m][k]
  __shared__ alignas(16) unsigned short Bs[128 * 32];   // [n][k] (transposed)

  const int tid = threadIdx.x;
  const int bz  = blockIdx.z;
  const int m0  = blockIdx.y * 128;
  const int n0  = blockIdx.x * 128;

  const size_t aoff = (size_t)bz * (size_t)strideA;
  const size_t boff = (size_t)bz * (size_t)strideB;
  const size_t coff = (size_t)bz * (size_t)strideC;

  const int wave = tid >> 5;
  const int lane = tid & 31;
  const int wm = (wave & 3) * 32;     // wave tile M offset
  const int wn = (wave >> 2) * 64;    // wave tile N offset
  const int r16 = lane & 15;
  const int hv  = lane >> 4;

  const bool fullM = (m0 + 128 <= M) && ((lda & 7) == 0);
  const bool fullN = (n0 + 128 <= N) && ((ldb & 7) == 0);

  v8f acc[2][4];
  for (int i = 0; i < 2; i++)
    for (int j = 0; j < 4; j++)
      for (int e = 0; e < 8; e++) acc[i][j][e] = 0.f;

  unsigned* As32w = (unsigned*)As;
  unsigned* Bs32w = (unsigned*)Bs;

  const int nK = K >> 5;
  for (int kb = 0; kb < nK; ++kb) {
    // ================= stage A tile =================
    if (!a_colmajor) {
      if (fullM) {
        // 128 rows x 64B: 2 x b128 per thread, async direct-to-LDS
        #pragma unroll
        for (int s = 0; s < 2; s++) {
          int seg = (tid << 1) | s;                 // 0..511
          int r = seg >> 2, part = (seg & 3) << 3;  // part: 0,8,16,24 elems
          const unsigned short* gp = &A[aoff + (size_t)(m0 + r) * lda + (kb << 5) + part];
          copy_b128_to_lds(gp, &As[r * 32 + part]);
        }
      } else {
        for (int e = tid; e < 4096; e += 256) {
          int r = e >> 5, c = e & 31;
          int gm = m0 + r, gk = (kb << 5) + c;
          unsigned short v = 0;
          if (gm < M && gk < K) v = A[aoff + (size_t)gm * lda + gk];
          As[r * 32 + c] = v;
        }
      }
    } else {
      if (fullM) {
        // pack k-pairs from two contiguous-in-m rows
        int kp = tid & 15, mg = tid >> 4;           // mg 0..15 -> m block of 8
        int k0 = (kb << 5) + (kp << 1);
        const unsigned short* g0 = &A[aoff + (size_t)k0 * lda + m0 + (mg << 3)];
        uint4 a0 = *(const uint4*)g0;
        uint4 a1 = *(const uint4*)(g0 + lda);
        const unsigned short* h0 = (const unsigned short*)&a0;
        const unsigned short* h1 = (const unsigned short*)&a1;
        #pragma unroll
        for (int e = 0; e < 8; e++)
          As32w[((mg << 3) + e) * 16 + kp] = (unsigned)h0[e] | ((unsigned)h1[e] << 16);
      } else {
        for (int e = tid; e < 4096; e += 256) {
          int r = e & 127, c = e >> 7;
          int gm = m0 + r, gk = (kb << 5) + c;
          unsigned short v = 0;
          if (gm < M && gk < K) v = A[aoff + (size_t)gk * lda + gm];
          As[r * 32 + c] = v;
        }
      }
    }
    // ================= stage B tile (transposed into [n][k]) =================
    if (b_is_f32) {
      const float* Bf = (const float*)Bsrc;
      if (fullN) {
        #pragma unroll
        for (int s = 0; s < 2; s++) {
          int idx = tid + (s << 8);                 // 0..511
          int kp = idx & 15, ng = idx >> 4;         // ng 0..31 -> n block of 4
          int k0 = (kb << 5) + (kp << 1);
          const float* g0 = &Bf[boff + (size_t)k0 * ldb + n0 + (ng << 2)];
          float4 f0 = *(const float4*)g0;
          float4 f1 = *(const float4*)(g0 + ldb);
          int nb = (ng << 2);
          Bs32w[(nb + 0) * 16 + kp] = (unsigned)f2bf(f0.x) | ((unsigned)f2bf(f1.x) << 16);
          Bs32w[(nb + 1) * 16 + kp] = (unsigned)f2bf(f0.y) | ((unsigned)f2bf(f1.y) << 16);
          Bs32w[(nb + 2) * 16 + kp] = (unsigned)f2bf(f0.z) | ((unsigned)f2bf(f1.z) << 16);
          Bs32w[(nb + 3) * 16 + kp] = (unsigned)f2bf(f0.w) | ((unsigned)f2bf(f1.w) << 16);
        }
      } else {
        for (int e = tid; e < 4096; e += 256) {
          int r = e & 127, c = e >> 7;
          int gn = n0 + r, gk = (kb << 5) + c;
          unsigned short v = 0;
          if (gn < N && gk < K) v = f2bf(Bf[boff + (size_t)gk * ldb + gn]);
          Bs[r * 32 + c] = v;
        }
      }
    } else {
      const unsigned short* Bh = (const unsigned short*)Bsrc;
      if (fullN) {
        int kp = tid & 15, ng = tid >> 4;           // ng 0..15 -> n block of 8
        int k0 = (kb << 5) + (kp << 1);
        const unsigned short* g0 = &Bh[boff + (size_t)k0 * ldb + n0 + (ng << 3)];
        uint4 a0 = *(const uint4*)g0;
        uint4 a1 = *(const uint4*)(g0 + ldb);
        const unsigned short* h0 = (const unsigned short*)&a0;
        const unsigned short* h1 = (const unsigned short*)&a1;
        #pragma unroll
        for (int e = 0; e < 8; e++)
          Bs32w[((ng << 3) + e) * 16 + kp] = (unsigned)h0[e] | ((unsigned)h1[e] << 16);
      } else {
        for (int e = tid; e < 4096; e += 256) {
          int r = e & 127, c = e >> 7;
          int gn = n0 + r, gk = (kb << 5) + c;
          unsigned short v = 0;
          if (gn < N && gk < K) v = Bh[boff + (size_t)gk * ldb + gn];
          Bs[r * 32 + c] = v;
        }
      }
    }
    // prefetch next B chunk (emits global_prefetch_b8)
    if (kb + 1 < nK) {
      const char* pf = (const char*)Bsrc;
      size_t elt = boff + (size_t)((kb << 5) + 32) * ldb + (size_t)n0 + (size_t)(tid & 127);
      __builtin_prefetch(pf + (elt << (b_is_f32 ? 2 : 1)), 0, 1);
    }
    staging_wait();
    __syncthreads();

    const unsigned* As32 = (const unsigned*)As;
    const unsigned* Bs32 = (const unsigned*)Bs;

    v8u afrag[2], bfrag[4];
    #pragma unroll
    for (int i = 0; i < 2; i++) {
      int row = wm + i * 16 + r16;
      #pragma unroll
      for (int j = 0; j < 8; j++) {
        int ko = (j < 4) ? (hv * 8 + 2 * j) : (16 + hv * 8 + 2 * (j - 4));
        afrag[i][j] = As32[(row * 32 + ko) >> 1];
      }
    }
    #pragma unroll
    for (int j4 = 0; j4 < 4; j4++) {
      int col = wn + j4 * 16 + r16;
      #pragma unroll
      for (int j = 0; j < 8; j++) {
        int ko = (j < 4) ? (hv * 8 + 2 * j) : (16 + hv * 8 + 2 * (j - 4));
        bfrag[j4][j] = Bs32[(col * 32 + ko) >> 1];
      }
    }
    #pragma unroll
    for (int i = 0; i < 2; i++) {
      v16bf av = __builtin_bit_cast(v16bf, afrag[i]);
      #pragma unroll
      for (int j = 0; j < 4; j++) {
        v16bf bv = __builtin_bit_cast(v16bf, bfrag[j]);
        acc[i][j] = __builtin_amdgcn_wmma_f32_16x16x32_bf16(
            false, av, false, bv, (short)0, acc[i][j], false, false);
      }
    }
    __syncthreads();
  }

  // ---- epilogue ----
  for (int i = 0; i < 2; i++) {
    for (int j = 0; j < 4; j++) {
      int gn = n0 + wn + j * 16 + r16;
      if (gn >= N) continue;
      #pragma unroll
      for (int r = 0; r < 8; r++) {
        int gm = m0 + wm + i * 16 + hv * 8 + r;
        if (gm >= M) continue;
        float v = acc[i][j][r] * alpha;
        if (bias) v += bias[gm];
        size_t o = coff + (size_t)gm * ldc + gn;
        if (C)   C[o] = v;
        if (Cbf) Cbf[o] = f2bf(v);
      }
    }
  }
}

// =====================================================================
// Dilated GLU conv as im2col GEMM: G[b][1024][N] = W[1024][2560] * im2col(H)
//   K index k -> ci = k/5, tap = k%5, input col = n + (tap-2)*dil
// =====================================================================
__global__ __launch_bounds__(256) void glu_conv_wmma(
    const unsigned short* __restrict__ W,     // [1024][2560] bf16 (layer base)
    const float* __restrict__ bias,           // [1024]
    const unsigned short* __restrict__ Hbf,   // [B][512][N] bf16
    float* __restrict__ G,                    // [B][1024][N]
    int dil, int N)
{
  __shared__ alignas(16) unsigned short As[128 * 32];
  __shared__ alignas(16) unsigned short Bs[128 * 32];

  const int tid = threadIdx.x;
  const int bz  = blockIdx.z;
  const int m0  = blockIdx.y * 128;
  const int n0  = blockIdx.x * 128;

  const unsigned short* Hb = Hbf + (size_t)bz * 512 * N;
  float* Gb = G + (size_t)bz * 1024 * N;

  const int wave = tid >> 5;
  const int lane = tid & 31;
  const int wm = (wave & 3) * 32;
  const int wn = (wave >> 2) * 64;
  const int r16 = lane & 15;
  const int hv  = lane >> 4;

  v8f acc[2][4];
  for (int i = 0; i < 2; i++)
    for (int j = 0; j < 4; j++)
      for (int e = 0; e < 8; e++) acc[i][j][e] = 0.f;

  const int nK = 2560 >> 5;  // 80
  for (int kb = 0; kb < nK; ++kb) {
    // A: weights, always full tiles -> async b128 direct to LDS
    #pragma unroll
    for (int s = 0; s < 2; s++) {
      int seg = (tid << 1) | s;
      int r = seg >> 2, part = (seg & 3) << 3;
      const unsigned short* gp = &W[(size_t)(m0 + r) * 2560 + (kb << 5) + part];
      copy_b128_to_lds(gp, &As[r * 32 + part]);
    }
    // B: im2col gather (shifted rows, edge-guarded) into [n][k]
    for (int e = tid; e < 4096; e += 256) {
      int r = e & 127, c = e >> 7;
      int k  = (kb << 5) + c;
      int ci = k / 5;
      int tap = k - ci * 5;
      int gn = n0 + r + (tap - 2) * dil;
      unsigned short v = 0;
      if (gn >= 0 && gn < N) v = Hb[(size_t)ci * N + gn];
      Bs[r * 32 + c] = v;
    }
    staging_wait();
    __syncthreads();

    const unsigned* As32 = (const unsigned*)As;
    const unsigned* Bs32 = (const unsigned*)Bs;
    v8u afrag[2], bfrag[4];
    #pragma unroll
    for (int i = 0; i < 2; i++) {
      int row = wm + i * 16 + r16;
      #pragma unroll
      for (int j = 0; j < 8; j++) {
        int ko = (j < 4) ? (hv * 8 + 2 * j) : (16 + hv * 8 + 2 * (j - 4));
        afrag[i][j] = As32[(row * 32 + ko) >> 1];
      }
    }
    #pragma unroll
    for (int j4 = 0; j4 < 4; j4++) {
      int col = wn + j4 * 16 + r16;
      #pragma unroll
      for (int j = 0; j < 8; j++) {
        int ko = (j < 4) ? (hv * 8 + 2 * j) : (16 + hv * 8 + 2 * (j - 4));
        bfrag[j4][j] = Bs32[(col * 32 + ko) >> 1];
      }
    }
    #pragma unroll
    for (int i = 0; i < 2; i++) {
      v16bf av = __builtin_bit_cast(v16bf, afrag[i]);
      #pragma unroll
      for (int j = 0; j < 4; j++) {
        v16bf bv = __builtin_bit_cast(v16bf, bfrag[j]);
        acc[i][j] = __builtin_amdgcn_wmma_f32_16x16x32_bf16(
            false, av, false, bv, (short)0, acc[i][j], false, false);
      }
    }
    __syncthreads();
  }

  for (int i = 0; i < 2; i++) {
    for (int j = 0; j < 4; j++) {
      int gn = n0 + wn + j * 16 + r16;
      if (gn >= N) continue;
      #pragma unroll
      for (int r = 0; r < 8; r++) {
        int gm = m0 + wm + i * 16 + hv * 8 + r;
        Gb[(size_t)gm * N + gn] = acc[i][j][r] + bias[gm];
      }
    }
  }
}

// ============================ elementwise ============================

__global__ void k_cvt_pad_cols(const float* __restrict__ src,
                               unsigned short* __restrict__ dst,
                               int R, int Cs, int Cd) {
  long long i = (long long)blockIdx.x * blockDim.x + threadIdx.x;
  if (i >= (long long)R * Cd) return;
  int r = (int)(i / Cd);
  int c = (int)(i - (long long)r * Cd);
  dst[i] = (c < Cs) ? f2bf(src[(size_t)r * Cs + c]) : (unsigned short)0;
}

__global__ void k_cvt_pad_rows_vec(const float* __restrict__ src,
                                   const float* __restrict__ vec,
                                   unsigned short* __restrict__ dst,
                                   int Bn, int Rs, int Rd, int Cols) {
  long long i = (long long)blockIdx.x * blockDim.x + threadIdx.x;
  if (i >= (long long)Bn * Rd * Cols) return;
  int c = (int)(i % Cols);
  long long t = i / Cols;
  int r = (int)(t % Rd);
  int b = (int)(t / Rd);
  float v = 0.f;
  if (r < Rs) {
    v = src[((size_t)b * Rs + r) * Cols + c];
    if (vec) v += vec[b * Rs + r];
  }
  dst[i] = f2bf(v);
}

__global__ void k_codevec(const float* __restrict__ ct, const float* __restrict__ W,
                          float* __restrict__ out, int Bn, int Dv) {
  int i = blockIdx.x * blockDim.x + threadIdx.x;
  if (i >= Bn * Dv) return;
  int b = i / Dv, j = i - b * Dv;
  float s = 0.f;
  #pragma unroll
  for (int c = 0; c < 8; c++) s += ct[b * 8 + c] * W[c * Dv + j];
  out[i] = s;
}

__global__ void k_glu(const float* __restrict__ G, float* __restrict__ H,
                      unsigned short* __restrict__ Hbf, long long total, int chstride) {
  long long idx = (long long)blockIdx.x * blockDim.x + threadIdx.x;
  if (idx >= total) return;
  int b = (int)(idx / chstride);
  long long rem = idx - (long long)b * chstride;
  const float* gb = G + (size_t)b * 2 * chstride;
  float a  = gb[rem];
  float gt = gb[rem + chstride];
  float h = (H[idx] + a * (1.f / (1.f + __expf(-gt)))) * SQRT_HALF_F;
  H[idx] = h;
  Hbf[idx] = f2bf(h);
}

__global__ void k_softmax(float* __restrict__ S, int Nn, int T) {
  int t = blockIdx.x * blockDim.x + threadIdx.x;
  int b = blockIdx.y;
  if (t >= T) return;
  float* base = S + (size_t)b * Nn * T + t;
  float mx = -3.4e38f;
  for (int n = 0; n < Nn; n++) mx = fmaxf(mx, base[(size_t)n * T]);
  float s = 0.f;
  for (int n = 0; n < Nn; n++) s += __expf(base[(size_t)n * T] - mx);
  float inv = 1.f / s;
  for (int n = 0; n < Nn; n++) {
    float v = __expf(base[(size_t)n * T] - mx) * inv;
    base[(size_t)n * T] = v;
  }
}

__global__ void k_concat(const float* __restrict__ R, const float* __restrict__ Q,
                         const float* __restrict__ vec, unsigned short* __restrict__ dst,
                         long long total) {
  long long idx = (long long)blockIdx.x * blockDim.x + threadIdx.x;
  if (idx >= total) return;
  int t  = (int)(idx % 2048);
  long long q = idx / 2048;
  int c2 = (int)(q % 1024);
  int b  = (int)(q / 1024);
  float v = (c2 < 512) ? R[((size_t)b * 512 + c2) * 2048 + t]
                       : Q[((size_t)b * 512 + (c2 - 512)) * 2048 + t];
  v += vec[b * 1024 + c2];
  dst[idx] = f2bf(v);
}

// =====================================================================
extern "C" void kernel_launch(void* const* d_in, const int* in_sizes, int n_in,
                              void* d_out, int out_size, void* d_ws, size_t ws_size,
                              hipStream_t stream) {
  (void)in_sizes; (void)n_in; (void)out_size; (void)ws_size;

  const float* in_s        = (const float*)d_in[0];
  const float* in_t        = (const float*)d_in[1];
  const float* c_t         = (const float*)d_in[2];
  const float* enc_start_w = (const float*)d_in[3];
  const float* enc_start_b = (const float*)d_in[4];
  const float* glu_w       = (const float*)d_in[5];
  const float* glu_b       = (const float*)d_in[6];
  const float* enc_end_w   = (const float*)d_in[7];
  const float* enc_end_b   = (const float*)d_in[8];
  const float* wc_pre      = (const float*)d_in[9];
  const float* w_pre       = (const float*)d_in[10];
  const float* b_pre       = (const float*)d_in[11];
  const float* wc_post     = (const float*)d_in[12];
  const float* w_post      = (const float*)d_in[13];
  const float* b_post      = (const float*)d_in[14];

  const int B = 16;

  char* ws = (char*)d_ws;
  size_t off = 0;
  auto alloc = [&](size_t bytes) -> void* {
    void* p = ws + off;
    off += (bytes + 255) & ~(size_t)255;
    return p;
  };

  unsigned short* Wsb = (unsigned short*)alloc((size_t)512 * 256 * 2);
  unsigned short* Wgb = (unsigned short*)alloc((size_t)8 * 1024 * 2560 * 2);
  unsigned short* Web = (unsigned short*)alloc((size_t)1024 * 512 * 2);
  unsigned short* Wpb = (unsigned short*)alloc((size_t)512 * 256 * 2);
  unsigned short* Wob = (unsigned short*)alloc((size_t)240 * 1024 * 2);
  unsigned short* Sbf = (unsigned short*)alloc((size_t)B * 256 * 2048 * 2);  // reused as t_bf
  float*          H   = (float*)alloc((size_t)B * 512 * 2048 * 4);           // reused as R
  unsigned short* Hbf = (unsigned short*)alloc((size_t)B * 512 * 2048 * 2);
  float*          G   = (float*)alloc((size_t)B * 1024 * 2048 * 4);          // reused as HRbf
  unsigned short* KVb = (unsigned short*)alloc((size_t)B * 1024 * 2048 * 2);
  float*          Q   = (float*)alloc((size_t)B * 512 * 2048 * 4);
  unsigned short* Qbf = (unsigned short*)alloc((size_t)B * 512 * 2048 * 2);
  float* cpre  = (float*)alloc((size_t)B * 240 * 4);
  float* cpost = (float*)alloc((size_t)B * 1024 * 4);
  unsigned short* HRbf = (unsigned short*)G;

  float* Yout = (float*)d_out;                              // [B,240,2048]
  float* Aout = (float*)d_out + (size_t)B * 240 * 2048;     // [B,2048,2048]

  dim3 blk(256);
  auto cdiv = [](long long a, long long b) { return (unsigned)((a + b - 1) / b); };

  // ---- weight conversions (fp32 -> bf16, K padded to mult of 32) ----
  k_cvt_pad_cols<<<cdiv((long long)512 * 256, 256), blk, 0, stream>>>(enc_start_w, Wsb, 512, 240, 256);
  k_cvt_pad_cols<<<cdiv((long long)8 * 1024 * 2560, 256), blk, 0, stream>>>(glu_w, Wgb, 8 * 1024, 2560, 2560);
  k_cvt_pad_cols<<<cdiv((long long)1024 * 512, 256), blk, 0, stream>>>(enc_end_w, Web, 1024, 512, 512);
  k_cvt_pad_cols<<<cdiv((long long)512 * 256, 256), blk, 0, stream>>>(w_pre, Wpb, 512, 240, 256);
  k_cvt_pad_cols<<<cdiv((long long)240 * 1024, 256), blk, 0, stream>>>(w_post, Wob, 240, 1024, 1024);

  // ---- encoder ----
  k_cvt_pad_rows_vec<<<cdiv((long long)B * 256 * 2048, 256), blk, 0, stream>>>(
      in_s, nullptr, Sbf, B, 240, 256, 2048);

  dim3 g512(16, 4, B);   // M=512
  gemm_wmma<<<g512, blk, 0, stream>>>(Wsb, (const void*)Sbf, enc_start_b, H, Hbf,
      512, 2048, 256, 256, 2048, 2048,
      0LL, (long long)256 * 2048, (long long)512 * 2048, 1.0f, 0, 0);

  for (int i = 0; i < 8; i++) {
    int dil = 1;
    for (int q = 0; q < (i % 4); q++) dil *= 3;
    dim3 gg(16, 8, B);   // M=1024
    glu_conv_wmma<<<gg, blk, 0, stream>>>(Wgb + (size_t)i * 1024 * 2560,
                                          glu_b + i * 1024, Hbf, G, dil, 2048);
    long long tot = (long long)B * 512 * 2048;
    k_glu<<<cdiv(tot, 256), blk, 0, stream>>>(G, H, Hbf, tot, 512 * 2048);
  }

  dim3 g1024(16, 8, B);  // M=1024
  gemm_wmma<<<g1024, blk, 0, stream>>>(Web, (const void*)Hbf, enc_end_b, nullptr, KVb,
      1024, 2048, 512, 512, 2048, 2048,
      0LL, (long long)512 * 2048, (long long)1024 * 2048, 1.0f, 0, 0);

  // ---- pre-decoder ----
  k_codevec<<<cdiv((long long)B * 240, 256), blk, 0, stream>>>(c_t, wc_pre, cpre, B, 240);
  k_codevec<<<cdiv((long long)B * 1024, 256), blk, 0, stream>>>(c_t, wc_post, cpost, B, 1024);

  k_cvt_pad_rows_vec<<<cdiv((long long)B * 256 * 2048, 256), blk, 0, stream>>>(
      in_t, cpre, Sbf, B, 240, 256, 2048);

  gemm_wmma<<<g512, blk, 0, stream>>>(Wpb, (const void*)Sbf, b_pre, Q, Qbf,
      512, 2048, 256, 256, 2048, 2048,
      0LL, (long long)256 * 2048, (long long)512 * 2048, 1.0f, 0, 0);

  // ---- attention: scores = K^T Q / sqrt(512) (A colmajor = K_s rows of KVb) ----
  dim3 gsc(16, 16, B);   // M=N=2048
  gemm_wmma<<<gsc, blk, 0, stream>>>(KVb, (const void*)Qbf, nullptr, Aout, nullptr,
      2048, 2048, 512, 2048, 2048, 2048,
      (long long)1024 * 2048, (long long)512 * 2048, (long long)2048 * 2048,
      0.044194173824159216f, 1, 0);

  dim3 gsm(2048 / 256, B);
  k_softmax<<<gsm, blk, 0, stream>>>(Aout, 2048, 2048);

  // ---- R = V * A  (V = KVb rows 512..1023; B = f32 attention in d_out) ----
  gemm_wmma<<<g512, blk, 0, stream>>>(KVb + (size_t)512 * 2048, (const void*)Aout, nullptr,
      H /* = R */, nullptr,
      512, 2048, 2048, 2048, 2048, 2048,
      (long long)1024 * 2048, (long long)2048 * 2048, (long long)512 * 2048,
      1.0f, 0, 1 /* b_is_f32 */);

  // ---- concat + code bias -> bf16 ----
  k_concat<<<cdiv((long long)B * 1024 * 2048, 256), blk, 0, stream>>>(
      H, Q, cpost, HRbf, (long long)B * 1024 * 2048);

  // ---- post conv: y = W_post @ HR + b_post ----
  dim3 gy(16, 2, B);     // M=240 (guarded)
  gemm_wmma<<<gy, blk, 0, stream>>>(Wob, (const void*)HRbf, b_post, Yout, nullptr,
      240, 2048, 1024, 1024, 2048, 2048,
      0LL, (long long)1024 * 2048, (long long)240 * 2048, 1.0f, 0, 0);
}
